// MyCrossAttention_1486058685018
// MI455X (gfx1250) — matrix operand order repaired and checked by reading, not compiled, not run on hardware
//
#include <hip/hip_runtime.h>
#include <hip/hip_bf16.h>
#include <math.h>

typedef __attribute__((ext_vector_type(16))) _Float16 v16h;
typedef __attribute__((ext_vector_type(8)))  float    v8f;

namespace {

constexpr int B_   = 4;
constexpr int C_   = 128;
constexpr int T_   = 400;
constexpr int F_   = 64;
constexpr int H_   = 4;
constexpr int HID_ = 32;
constexpr int NBLK = B_ * T_;      // 1600 workgroups, one per (b, t)
constexpr int NCOL = 2 * F_;       // 128 columns: col = s*64 + f
constexpr float EPS = 1e-5f;

constexpr int LDH = 136;           // f16 leading dim (K direction), padded vs 128
constexpr int LDF = 132;           // f32 leading dim, padded vs 128

// dynamic LDS carve (bytes)
constexpr int OFF_WB = 0;                         // weights, f16 [128][LDH]   = 34816 B
constexpr int OFF_XT = OFF_WB + C_ * LDH * 2;     // X^T,     f16 [128][LDH]   = 34816 B
constexpr int OFF_QB = OFF_XT + NCOL * LDH * 2;   // Q,       f32 [128][LDF]   = 67584 B (later aliased by O^T f16)
constexpr int OFF_KV = OFF_QB + C_ * LDF * 4;     // K then V, f32 [128][LDF]  = 67584 B (later aliased by Y f32)
constexpr int SMEM_BYTES = OFF_KV + C_ * LDF * 4; // 204800 B  (<= 320 KB WGP LDS)
static_assert(SMEM_BYTES <= 320 * 1024, "LDS budget");

__device__ inline float waveReduceAdd(float v) {
#pragma unroll
  for (int off = 16; off > 0; off >>= 1) v += __shfl_xor(v, off, 32);
  return v;
}

// Load a 16x32 f16 fragment from row-major LDS [idx][k] (ld = LDH).
// isA: A-matrix lane layout (lanes 0-15: K pairs 0,2,4,6,16,18,20,22; lanes 16-31: +8)
// !isA: B-matrix layout from K-major storage (same K order as A)
__device__ inline v16h load_frag(const _Float16* __restrict__ base, int idx, int k0,
                                 int lane, bool isA) {
  union { v16h v; unsigned u[8]; } fr;
  const _Float16* p = base + idx * LDH + k0;
#pragma unroll
  for (int j = 0; j < 8; ++j) {
    const int kb = ((j < 4) ? (2 * j) : (16 + 2 * (j - 4))) + ((lane & 16) ? 8 : 0);
    fr.u[j] = *(const unsigned*)(p + kb);            // two packed halves (K, K+1)
  }
  return fr.v;
}

// C[128x128] = A[128x128] * B[128x128], A row-major f16 LDS, B stored transposed
// ([n][k], K-major) f16 LDS. Output f32 to LDS with bias + PReLU fused.
__device__ inline void gemm128(const _Float16* __restrict__ A, const _Float16* __restrict__ Bt,
                               float* __restrict__ Cout,
                               const float* __restrict__ bias, const float* __restrict__ prelu,
                               int pdiv, int wave, int lane) {
  for (int tile = wave; tile < 64; tile += 8) {       // 8x8 grid of 16x16 tiles, 8 waves
    const int mT = tile >> 3, nT = tile & 7;
    const int mrow = mT * 16 + (lane & 15);
    const int ncol = nT * 16 + (lane & 15);
    v8f acc = {};
#pragma unroll
    for (int k0 = 0; k0 < 128; k0 += 32) {
      v16h a = load_frag(A,  mrow, k0, lane, true);
      v16h b = load_frag(Bt, ncol, k0, lane, false);
      acc = __builtin_amdgcn_wmma_f32_16x16x32_f16(false, a, false, b,
                                                   (short)0, acc, false, false);
    }
    const int rbase = mT * 16 + ((lane & 16) ? 8 : 0);
#pragma unroll
    for (int r = 0; r < 8; ++r) {
      const int row = rbase + r;
      float v = acc[r] + bias[row];
      const float pp = prelu[row / pdiv];
      v = (v >= 0.f) ? v : pp * v;
      Cout[row * LDF + ncol] = v;
    }
  }
}

// LayerNorm over one (head, stream) 32x64 block; one wave per block (8 blocks, 8 waves).
__device__ inline void ln_head(float* __restrict__ Buf, const float* __restrict__ g,
                               const float* __restrict__ z, int wave, int lane) {
  const int h = wave >> 1, s = wave & 1;
  float sum = 0.f, sq = 0.f;
  for (int i = lane; i < HID_ * F_; i += 32) {
    const int o = i >> 6, f = i & 63;
    const float v = Buf[(h * HID_ + o) * LDF + s * F_ + f];
    sum += v; sq += v * v;
  }
  sum = waveReduceAdd(sum);
  sq  = waveReduceAdd(sq);
  const float mu = sum * (1.f / (HID_ * F_));
  const float rs = rsqrtf(sq * (1.f / (HID_ * F_)) - mu * mu + EPS);
  for (int i = lane; i < HID_ * F_; i += 32) {
    const int o = i >> 6, f = i & 63;
    float* p = &Buf[(h * HID_ + o) * LDF + s * F_ + f];
    const int gi = (h * HID_ + o) * F_ + f;
    *p = (*p - mu) * rs * g[gi] + z[gi];
  }
}

__global__ __launch_bounds__(256)
void fused_cross_attn(const float* __restrict__ x1, const float* __restrict__ x2,
                      const float* __restrict__ Wq, const float* __restrict__ bq,
                      const float* __restrict__ pq, const float* __restrict__ gq,
                      const float* __restrict__ zq,
                      const float* __restrict__ Wk, const float* __restrict__ bk,
                      const float* __restrict__ pk, const float* __restrict__ gk,
                      const float* __restrict__ zk,
                      const float* __restrict__ Wv, const float* __restrict__ bv,
                      const float* __restrict__ pv, const float* __restrict__ gv,
                      const float* __restrict__ zv,
                      const float* __restrict__ Wo, const float* __restrict__ bo,
                      const float* __restrict__ po, const float* __restrict__ go,
                      const float* __restrict__ zo,
                      float* __restrict__ out) {
  extern __shared__ char smem[];
  _Float16* Wb = (_Float16*)(smem + OFF_WB);
  _Float16* Xt = (_Float16*)(smem + OFF_XT);
  float*    Qb = (float*)   (smem + OFF_QB);
  float*    KV = (float*)   (smem + OFF_KV);
  _Float16* Ot = (_Float16*)(smem + OFF_QB);   // O^T aliases Q after scores are done
  float*    Yb = (float*)   (smem + OFF_KV);   // Y aliases V after the attn mix

  __shared__ float sAttn[H_][2][2];
  __shared__ float sPart[8][4];

  const int tid  = threadIdx.x;
  const int lane = tid & 31;
  const int wave = tid >> 5;

  const int n = blockIdx.x;
  const int b = n / T_;
  const int t = n - b * T_;
  const size_t xbase = (size_t)b * C_ * T_ * F_ + (size_t)t * F_;

  auto stageW = [&](const float* Wsrc) {   // f32 [128][128] -> f16 LDS [row][k]
    for (int i = tid; i < C_ * C_; i += 256)
      Wb[(i >> 7) * LDH + (i & 127)] = (_Float16)Wsrc[i];
  };

  // ---- stage X^T (col-major wrt B: [n = s*64+f][k = c]) and Wq ----
  for (int i = tid; i < C_ * F_; i += 256) {
    const int c = i >> 6, f = i & 63;
    const size_t g = xbase + (size_t)c * T_ * F_ + f;
    Xt[f * LDH + c]        = (_Float16)x1[g];
    Xt[(F_ + f) * LDH + c] = (_Float16)x2[g];
  }
  stageW(Wq);
  __syncthreads();

  // ---- Q = Wq_all @ X  (+bias, PReLU) ----
  gemm128(Wb, Xt, Qb, bq, pq, HID_, wave, lane);
  __syncthreads();

  // ---- LN(Q) per (h,s); stage Wk ----
  ln_head(Qb, gq, zq, wave, lane);
  stageW(Wk);
  __syncthreads();

  // ---- K = Wk_all @ X ----
  gemm128(Wb, Xt, KV, bk, pk, HID_, wave, lane);
  __syncthreads();

  // ---- LN(K); stage Wv ----
  ln_head(KV, gk, zk, wave, lane);
  stageW(Wv);
  __syncthreads();

  // ---- attention scores + softmax (d = HID*F = 2048, 2x2 per head) ----
  if (wave < H_) {
    const int h = wave;
    float d00 = 0.f, d01 = 0.f, d10 = 0.f, d11 = 0.f;
    for (int i = lane; i < HID_ * F_; i += 32) {
      const int o = i >> 6, f = i & 63;
      const float* qr = &Qb[(h * HID_ + o) * LDF];
      const float* kr = &KV[(h * HID_ + o) * LDF];
      const float q0 = qr[f], q1 = qr[F_ + f];
      const float k0 = kr[f], k1 = kr[F_ + f];
      d00 += q0 * k0; d01 += q0 * k1; d10 += q1 * k0; d11 += q1 * k1;
    }
    d00 = waveReduceAdd(d00); d01 = waveReduceAdd(d01);
    d10 = waveReduceAdd(d10); d11 = waveReduceAdd(d11);
    if (lane == 0) {
      const float scale = 0.02209708691f;  // 1/sqrt(2048)
      const float r00 = d00 * scale, r01 = d01 * scale;
      const float r10 = d10 * scale, r11 = d11 * scale;
      const float m0 = fmaxf(r00, r01);
      const float e00 = __expf(r00 - m0), e01 = __expf(r01 - m0);
      const float i0 = 1.f / (e00 + e01);
      sAttn[h][0][0] = e00 * i0; sAttn[h][0][1] = e01 * i0;
      const float m1 = fmaxf(r10, r11);
      const float e10 = __expf(r10 - m1), e11 = __expf(r11 - m1);
      const float i1 = 1.f / (e10 + e11);
      sAttn[h][1][0] = e10 * i1; sAttn[h][1][1] = e11 * i1;
    }
  }
  __syncthreads();

  // ---- V = Wv_all @ X  (overwrites K, scores already consumed it) ----
  gemm128(Wb, Xt, KV, bv, pv, HID_, wave, lane);
  __syncthreads();

  // ---- LN(V) ----
  ln_head(KV, gv, zv, wave, lane);
  __syncthreads();

  // ---- O = attn-mix of V columns; store transposed f16 (overwrites Q) ----
  for (int i = tid; i < C_ * NCOL; i += 256) {
    const int row = i >> 7, col = i & 127;      // row = h*32+cv, col = s*64+f
    const int h = row >> 5, s = col >> 6, f = col & 63;
    const float a0 = sAttn[h][s][0], a1 = sAttn[h][s][1];
    const float ov = a0 * KV[row * LDF + f] + a1 * KV[row * LDF + F_ + f];
    Ot[col * LDH + row] = (_Float16)ov;
  }
  __syncthreads();
  stageW(Wo);
  __syncthreads();

  // ---- Y = Wo @ O (+bias, PReLU); overwrites V buffer ----
  gemm128(Wb, Ot, Yb, bo, po, C_, wave, lane);
  __syncthreads();

  // ---- final LN over (C, F) per stream: cross-wave reduction ----
  float s0 = 0.f, q0 = 0.f, s1 = 0.f, q1 = 0.f;
  for (int i = tid; i < C_ * F_; i += 256) {
    const int o = i >> 6, f = i & 63;
    const float v0 = Yb[o * LDF + f];
    const float v1 = Yb[o * LDF + F_ + f];
    s0 += v0; q0 += v0 * v0; s1 += v1; q1 += v1 * v1;
  }
  s0 = waveReduceAdd(s0); q0 = waveReduceAdd(q0);
  s1 = waveReduceAdd(s1); q1 = waveReduceAdd(q1);
  if (lane == 0) {
    sPart[wave][0] = s0; sPart[wave][1] = q0;
    sPart[wave][2] = s1; sPart[wave][3] = q1;
  }
  __syncthreads();
  float ts0 = 0.f, tq0 = 0.f, ts1 = 0.f, tq1 = 0.f;
#pragma unroll
  for (int w = 0; w < 8; ++w) {
    ts0 += sPart[w][0]; tq0 += sPart[w][1];
    ts1 += sPart[w][2]; tq1 += sPart[w][3];
  }
  const float inv = 1.f / (C_ * F_);
  const float mu0 = ts0 * inv, mu1 = ts1 * inv;
  const float rs0 = rsqrtf(tq0 * inv - mu0 * mu0 + EPS);
  const float rs1 = rsqrtf(tq1 * inv - mu1 * mu1 + EPS);

  // ---- affine + residual + store (y1 then y2, flat concat) ----
  float* out1 = out;
  float* out2 = out + (size_t)B_ * C_ * T_ * F_;
  for (int i = tid; i < C_ * F_; i += 256) {
    const int o = i >> 6, f = i & 63;
    const int gi = o * F_ + f;
    const float gg = go[gi], zz = zo[gi];
    const float v0 = (Yb[o * LDF + f]      - mu0) * rs0 * gg + zz;
    const float v1 = (Yb[o * LDF + F_ + f] - mu1) * rs1 * gg + zz;
    const size_t g = xbase + (size_t)o * T_ * F_ + f;
    out1[g] = v0 + x1[g];
    out2[g] = v1 + x2[g];
  }
}

} // anonymous namespace

extern "C" void kernel_launch(void* const* d_in, const int* in_sizes, int n_in,
                              void* d_out, int out_size, void* d_ws, size_t ws_size,
                              hipStream_t stream) {
  (void)in_sizes; (void)n_in; (void)out_size; (void)d_ws; (void)ws_size;
  const float* p[22];
  for (int i = 0; i < 22; ++i) p[i] = (const float*)d_in[i];

  hipFuncSetAttribute(reinterpret_cast<const void*>(&fused_cross_attn),
                      hipFuncAttributeMaxDynamicSharedMemorySize, SMEM_BYTES);

  fused_cross_attn<<<NBLK, 256, SMEM_BYTES, stream>>>(
      p[0], p[1],
      p[2], p[3], p[4], p[5], p[6],
      p[7], p[8], p[9], p[10], p[11],
      p[12], p[13], p[14], p[15], p[16],
      p[17], p[18], p[19], p[20], p[21],
      (float*)d_out);
}
